// DNDLSTMCell_47631187312927
// MI455X (gfx1250) — matrix operand order repaired and checked by reading, not compiled, not run on hardware
//
#include <hip/hip_runtime.h>

// DND-LSTM cell for MI455X (gfx1250), wave32, bf16 WMMA with f32 accumulate.
// B=256, D_IN=256, H=256, G=1280, L=100000.
// Workspace (floats): preact[256*1280] | part_acc[25*256*256] | part_m[25*256] | part_l[25*256]
// Total ws use ~7.93 MB.

#define EPSV 1e-8f

typedef __attribute__((ext_vector_type(16))) __bf16 v16bf;
typedef __attribute__((ext_vector_type(8)))  float  v8f;

union FragU { uint4 q[2]; unsigned u[8]; v16bf v; };

__device__ __forceinline__ unsigned short f2bf(float f) {
  // round-to-nearest-even f32 -> bf16
  unsigned x = __builtin_bit_cast(unsigned, f);
  unsigned r = x + 0x7FFFu + ((x >> 16) & 1u);
  return (unsigned short)(r >> 16);
}
__device__ __forceinline__ unsigned pack2(float a, float b) {
  return (unsigned)f2bf(a) | ((unsigned)f2bf(b) << 16);
}
__device__ __forceinline__ float sigm(float x) { return 1.f / (1.f + __expf(-x)); }

// A-fragment (16x32 bf16, ISA 7.12.2): lane m = row m, VGPR i<4 -> pair i+kh*4,
// VGPR i>=4 -> pair 8+(i-4)+kh*4. Two contiguous 16B runs per lane.
__device__ __forceinline__ v16bf load_afrag(const unsigned* rowBase, int pb, int kh) {
  FragU f;
  const uint4* r = (const uint4*)(rowBase + pb);   // pb multiple of 4, base 16B-aligned
  f.q[0] = r[kh];                                  // pairs pb + kh*4     .. +3
  f.q[1] = r[2 + kh];                              // pairs pb + 8 + kh*4 .. +3
  return f.v;
}
// B-fragment (32x16 bf16): lane n = column n, VGPR i<4 -> pair i+kh*4,
// VGPR i>=4 -> pair 8+(i-4)+kh*4. Two contiguous 16B runs per lane.
__device__ __forceinline__ v16bf load_bfrag(const unsigned* colBase, int pb, int kh) {
  FragU f;
  const uint4* r = (const uint4*)(colBase + pb);
  f.q[0] = r[kh];                                  // pairs pb + kh*4     .. +3
  f.q[1] = r[2 + kh];                              // pairs pb + 8 + kh*4 .. +3
  return f.v;
}

// ---------------------------------------------------------------------------
// Phase A: preact = [x_t | h] @ [W_i2h | W_h2h]^T   (256 x 1280, K = 512)
// 64x64 C-tile per 128-thread workgroup, bf16 WMMA, biases added in combine.
// ---------------------------------------------------------------------------
__global__ __launch_bounds__(128)
void preact_gemm_kernel(const float* __restrict__ x_t, const float* __restrict__ h,
                        const float* __restrict__ Wi,  const float* __restrict__ Wh,
                        float* __restrict__ pre) {
  __shared__ __align__(16) unsigned Apack[64][20];   // [row][d-pair], 80B row stride
  __shared__ __align__(16) unsigned Bpack[64][20];   // [g-col][d-pair]

  const int t  = threadIdx.x;
  const int w  = t >> 5;
  const int ln = t & 31;
  const int m  = ln & 15;
  const int kh = ln >> 4;
  const int rbase = blockIdx.x * 64;
  const int g0    = blockIdx.y * 64;

  v8f zero = {0.f,0.f,0.f,0.f,0.f,0.f,0.f,0.f};
  v8f accv[4];
  #pragma unroll
  for (int i = 0; i < 4; ++i) accv[i] = zero;

  for (int c = 0; c < 16; ++c) {          // K chunks of 32 (never straddles x/h or Wi/Wh)
    const int d0 = c * 32;
    __syncthreads();
    { // A: concat(x,h) rows -> bf16 pairs; 16 consecutive floats per thread
      int row = t >> 1, p0 = (t & 1) * 8, d = d0 + 2 * p0;
      const float4* sv = (const float4*)((d < 256)
          ? (x_t + (size_t)(rbase + row) * 256 + d)
          : (h   + (size_t)(rbase + row) * 256 + d - 256));
      uint2* arow = (uint2*)&Apack[row][p0];
      #pragma unroll
      for (int q = 0; q < 4; ++q) {
        float4 v = sv[q];
        arow[q] = make_uint2(pack2(v.x, v.y), pack2(v.z, v.w));
      }
    }
    { // B: Wc^T stored [col][pair]; 16 consecutive floats of one W row per thread
      int col = t >> 1, p0 = (t & 1) * 8, d = d0 + 2 * p0;
      int g = g0 + col;
      const float4* sv = (const float4*)((d < 256)
          ? (Wi + (size_t)g * 256 + d)
          : (Wh + (size_t)g * 256 + d - 256));
      uint2* brow = (uint2*)&Bpack[col][p0];
      #pragma unroll
      for (int q = 0; q < 4; ++q) {
        float4 v = sv[q];
        brow[q] = make_uint2(pack2(v.x, v.y), pack2(v.z, v.w));
      }
    }
    __syncthreads();

    v16bf af = load_afrag(&Apack[w * 16 + m][0], 0, kh);
    #pragma unroll
    for (int tt = 0; tt < 4; ++tt) {
      v16bf bv = load_bfrag(&Bpack[tt * 16 + m][0], 0, kh);
      accv[tt] = __builtin_amdgcn_wmma_f32_16x16x32_bf16(false, af, false, bv,
                                                         (short)0, accv[tt], false, false);
    }
  }

  #pragma unroll
  for (int j = 0; j < 8; ++j) {
    int row = rbase + w * 16 + j + 8 * kh;           // C-frag row mapping
    #pragma unroll
    for (int tt = 0; tt < 4; ++tt)
      pre[(size_t)row * 1280 + g0 + tt * 16 + m] = accv[tt][j];
  }
}

// ---------------------------------------------------------------------------
// Phase B: flash-style DND attention partials.
// grid = (4 query blocks of 64 rows) x (25 L-slices of 4000 keys); slice-major
// launch order so the 4 consumers of each 8.2MB key/val slice overlap in the
// 192MB L2 -> HBM traffic stays near the 205MB floor (~9us @ 23.3TB/s).
// Each wave: 16 query rows x full H=256, online softmax in f32.
// ---------------------------------------------------------------------------
__global__ __launch_bounds__(128)
void dnd_attention_kernel(const float* __restrict__ x_t,
                          const float* __restrict__ keys,
                          const float* __restrict__ vals,
                          float* __restrict__ part_acc,
                          float* __restrict__ part_m,
                          float* __restrict__ part_l) {
  __shared__ __align__(16) unsigned Qpack[64][132];   // [q-row][d-pair] normalized bf16
  __shared__ __align__(16) unsigned KK[32][132];      // [key][d-pair]
  __shared__ __align__(16) unsigned VV[256][20];      // [h][key-pair]
  __shared__ __align__(16) unsigned Ppack[4][16][20]; // per-wave P tile [row][key-pair]
  __shared__ float kn2p[32][4];                       // per-key sum-of-squares partials
  __shared__ float qscale[64];

  const int t  = threadIdx.x;
  const int w  = t >> 5;
  const int ln = t & 31;
  const int m  = ln & 15;
  const int kh = ln >> 4;

  const int qbase = blockIdx.x * 64;
  const int slice = blockIdx.y;
  const int sbase = slice * 4000;

  // ---- stage normalized queries once ----
  if (t < 64) {
    const float4* xr = (const float4*)(x_t + (size_t)(qbase + t) * 256);
    float ss = 0.f;
    for (int q = 0; q < 64; ++q) {
      float4 v = xr[q];
      ss += v.x*v.x + v.y*v.y + v.z*v.z + v.w*v.w;
    }
    qscale[t] = 1.f / fmaxf(sqrtf(ss), EPSV);
  }
  __syncthreads();
  { // pack: thread t covers row t/2, 64 consecutive d-pairs
    int row = t >> 1, p0 = (t & 1) * 64;
    float s = qscale[row];
    const float4* xr = (const float4*)(x_t + (size_t)(qbase + row) * 256 + 2 * p0);
    uint2* qrow = (uint2*)&Qpack[row][p0];
    #pragma unroll
    for (int q = 0; q < 32; ++q) {
      float4 v = xr[q];
      qrow[q] = make_uint2(pack2(v.x * s, v.y * s), pack2(v.z * s, v.w * s));
    }
  }

  v8f zero = {0.f,0.f,0.f,0.f,0.f,0.f,0.f,0.f};
  v8f acc[16];                            // 16 rows x 256 H, f32
  float mrow[8], lrow[8];
  #pragma unroll
  for (int i = 0; i < 16; ++i) acc[i] = zero;
  #pragma unroll
  for (int j = 0; j < 8; ++j) { mrow[j] = -1e30f; lrow[j] = 0.f; }

  for (int tile = 0; tile < 125; ++tile) {
    const int kbase = sbase + tile * 32;
    __syncthreads();
    { // stage 32 keys -> KK[key][d-pair] (contiguous uint2 stores) + norms
      const int kk = t >> 2;
      const int d0 = (t & 3) * 64;
      const float4* kr = (const float4*)(keys + (size_t)(kbase + kk) * 256 + d0);
      uint2* krow = (uint2*)&KK[kk][d0 >> 1];
      float ss = 0.f;
      #pragma unroll
      for (int q = 0; q < 16; ++q) {
        float4 v = kr[q];
        ss += v.x*v.x + v.y*v.y + v.z*v.z + v.w*v.w;
        krow[q] = make_uint2(pack2(v.x, v.y), pack2(v.z, v.w));
      }
      kn2p[kk][t & 3] = ss;
      if (tile + 1 < 125)
        __builtin_prefetch(keys + (size_t)(kbase + 32 + kk) * 256 + d0, 0, 1);
    }
    { // stage 32 vals -> VV[h][key-pair] (key-pair packed bf16)
      const int p  = t >> 3;
      const int h0 = (t & 7) * 32;
      const float4* r0 = (const float4*)(vals + (size_t)(kbase + 2*p)     * 256 + h0);
      const float4* r1 = (const float4*)(vals + (size_t)(kbase + 2*p + 1) * 256 + h0);
      #pragma unroll
      for (int q = 0; q < 8; ++q) {
        float4 a = r0[q]; float4 b = r1[q];
        int hh = h0 + q * 4;
        VV[hh][p]     = pack2(a.x, b.x);
        VV[hh + 1][p] = pack2(a.y, b.y);
        VV[hh + 2][p] = pack2(a.z, b.z);
        VV[hh + 3][p] = pack2(a.w, b.w);
      }
      if (tile + 1 < 125)
        __builtin_prefetch(vals + (size_t)(kbase + 32 + 2*p) * 256 + h0, 0, 1);
    }
    __syncthreads();

    // ---- scores: S(16q x 32k) = Qn(16x256) x K^T(256x32), 16 WMMAs ----
    const unsigned* qrow  = &Qpack[w * 16 + m][0];
    const unsigned* k0row = &KK[m][0];
    const unsigned* k1row = &KK[16 + m][0];
    v8f S0 = zero, S1 = zero;
    #pragma unroll
    for (int c = 0; c < 8; ++c) {
      const int pb = c * 16;
      v16bf af = load_afrag(qrow, pb, kh);
      v16bf b0 = load_bfrag(k0row, pb, kh);
      v16bf b1 = load_bfrag(k1row, pb, kh);
      S0 = __builtin_amdgcn_wmma_f32_16x16x32_bf16(false, af, false, b0, (short)0, S0, false, false);
      S1 = __builtin_amdgcn_wmma_f32_16x16x32_bf16(false, af, false, b1, (short)0, S1, false, false);
    }

    // fold in key norms: dot(qn, k/||k||) = dot(qn,k)/max(||k||,eps)
    float n0 = kn2p[m][0] + kn2p[m][1] + kn2p[m][2] + kn2p[m][3];
    float n1 = kn2p[16 + m][0] + kn2p[16 + m][1] + kn2p[16 + m][2] + kn2p[16 + m][3];
    float sc0 = 1.f / fmaxf(sqrtf(n0), EPSV);
    float sc1 = 1.f / fmaxf(sqrtf(n1), EPSV);

    // ---- online softmax per row (wave halves hold rows j and j+8) ----
    float alpha[8];
    #pragma unroll
    for (int j = 0; j < 8; ++j) {
      float s0 = S0[j] * sc0;
      float s1 = S1[j] * sc1;
      float mx = fmaxf(s0, s1);
      #pragma unroll
      for (int o = 8; o > 0; o >>= 1) mx = fmaxf(mx, __shfl_xor(mx, o, 16));
      float mn = fmaxf(mrow[j], mx);
      float al = __expf(mrow[j] - mn);
      float p0 = __expf(s0 - mn);
      float p1 = __expf(s1 - mn);
      float ps = p0 + p1;
      #pragma unroll
      for (int o = 8; o > 0; o >>= 1) ps += __shfl_xor(ps, o, 16);
      lrow[j] = lrow[j] * al + ps;
      mrow[j] = mn;
      alpha[j] = al;
      // write P into A-fragment pair layout (adjacent keys packed via shfl)
      float q0 = __shfl_xor(p0, 1, 16);
      float q1 = __shfl_xor(p1, 1, 16);
      if ((m & 1) == 0) {
        Ppack[w][j + 8 * kh][m >> 1]       = pack2(p0, q0);
        Ppack[w][j + 8 * kh][8 + (m >> 1)] = pack2(p1, q1);
      }
    }
    #pragma unroll
    for (int tt = 0; tt < 16; ++tt) {
      #pragma unroll
      for (int j = 0; j < 8; ++j) acc[tt][j] *= alpha[j];
    }

    // ---- acc += P(16x32) @ V(32x256), 16 WMMAs ----
    v16bf pf = load_afrag(&Ppack[w][m][0], 0, kh);
    #pragma unroll
    for (int tt = 0; tt < 16; ++tt) {
      v16bf bv = load_bfrag(&VV[tt * 16 + m][0], 0, kh);
      acc[tt] = __builtin_amdgcn_wmma_f32_16x16x32_bf16(false, pf, false, bv,
                                                        (short)0, acc[tt], false, false);
    }
  }

  // ---- write slice partials ----
  #pragma unroll
  for (int j = 0; j < 8; ++j) {
    const int row = qbase + w * 16 + j + 8 * kh;
    float* dst = part_acc + ((size_t)slice * 256 + row) * 256;
    #pragma unroll
    for (int tt = 0; tt < 16; ++tt) dst[tt * 16 + m] = acc[tt][j];
    if (m == 0) {
      part_m[slice * 256 + row] = mrow[j];
      part_l[slice * 256 + row] = lrow[j];
    }
  }
}

// ---------------------------------------------------------------------------
// Phase C: merge 25 slice partials, tanh, and all LSTM gate math.
// ---------------------------------------------------------------------------
__global__ __launch_bounds__(256)
void combine_kernel(const float* __restrict__ c_in,
                    const float* __restrict__ bi, const float* __restrict__ bh,
                    const float* __restrict__ pre,
                    const float* __restrict__ pacc,
                    const float* __restrict__ pm, const float* __restrict__ pl,
                    float* __restrict__ out) {
  const int b = blockIdx.x;
  const int hh = threadIdx.x;
  __shared__ float e[25];
  __shared__ float gl_s;
  if (hh == 0) {
    float gm = -1e30f;
    for (int s = 0; s < 25; ++s) gm = fmaxf(gm, pm[s * 256 + b]);
    float gl = 0.f;
    for (int s = 0; s < 25; ++s) {
      float ee = __expf(pm[s * 256 + b] - gm);
      e[s] = ee;
      gl += pl[s * 256 + b] * ee;
    }
    gl_s = gl;
  }
  __syncthreads();
  float a = 0.f;
  for (int s = 0; s < 25; ++s)
    a += pacc[((size_t)s * 256 + b) * 256 + hh] * e[s];
  float mval = tanhf(a / gl_s);

  const float* pr = pre + (size_t)b * 1280;
  float f  = sigm(pr[hh]         + bi[hh]         + bh[hh]);
  float it = sigm(pr[256 + hh]   + bi[256 + hh]   + bh[256 + hh]);
  float o  = sigm(pr[512 + hh]   + bi[512 + hh]   + bh[512 + hh]);
  float r  = sigm(pr[768 + hh]   + bi[768 + hh]   + bh[768 + hh]);
  float cn = tanhf(pr[1024 + hh] + bi[1024 + hh]  + bh[1024 + hh]);
  float ct = f * c_in[b * 256 + hh] + it * cn + r * mval;
  out[b * 256 + hh]         = o * tanhf(ct);   // h_t
  out[65536 + b * 256 + hh] = ct;              // c_t
}

extern "C" void kernel_launch(void* const* d_in, const int* in_sizes, int n_in,
                              void* d_out, int out_size, void* d_ws, size_t ws_size,
                              hipStream_t stream) {
  (void)in_sizes; (void)n_in; (void)out_size; (void)ws_size;
  const float* x_t  = (const float*)d_in[0];
  const float* h    = (const float*)d_in[1];
  const float* c    = (const float*)d_in[2];
  const float* Wi   = (const float*)d_in[3];
  const float* bi   = (const float*)d_in[4];
  const float* Wh   = (const float*)d_in[5];
  const float* bh   = (const float*)d_in[6];
  const float* keys = (const float*)d_in[7];
  const float* vals = (const float*)d_in[8];
  float* out = (float*)d_out;

  float* ws   = (float*)d_ws;
  float* pre  = ws;                    // 256*1280   = 327680 f
  float* pacc = pre + 327680;          // 25*256*256 = 1638400 f
  float* pm   = pacc + 1638400;        // 25*256     = 6400 f
  float* pl   = pm + 6400;             // 25*256     = 6400 f

  preact_gemm_kernel<<<dim3(4, 20), 128, 0, stream>>>(x_t, h, Wi, Wh, pre);
  dnd_attention_kernel<<<dim3(4, 25), 128, 0, stream>>>(x_t, keys, vals, pacc, pm, pl);
  combine_kernel<<<256, 256, 0, stream>>>(c, bi, bh, pre, pacc, pm, pl, out);
}
